// DeformableConv2d_78340203479506
// MI455X (gfx1250) — compile-verified, matrix operand order checked by reading, not run on hardware
//
#include <hip/hip_runtime.h>
#include <hip/hip_bf16.h>

typedef __attribute__((ext_vector_type(16))) _Float16 v16h;
typedef __attribute__((ext_vector_type(8)))  _Float16 v8h;
typedef __attribute__((ext_vector_type(8)))  float    v8f;

#define BB 8
#define CC 64
#define OO 64
#define HH 128
#define WW 128
#define KPNT 9
#define KDIM (CC * KPNT)      // 576
#define TILE_M 32             // pixels per block (one row strip)
#define NCHUNK (KDIM / 32)    // 18 WMMA K-steps

// ---------------------------------------------------------------------------
// Kernel 1: repack weight [O][C][3][3] (f32) -> wt[o][k] (f16), k = p*64 + c.
// 73728 bytes into d_ws; L2-resident for the whole main kernel.
// ---------------------------------------------------------------------------
__global__ __launch_bounds__(256) void dcn_prep_weights(const float* __restrict__ w,
                                                        _Float16* __restrict__ wt) {
    int i = blockIdx.x * 256 + threadIdx.x;
    if (i >= OO * KDIM) return;
    int o = i / KDIM;
    int k = i - o * KDIM;
    int p = k >> 6;          // kernel point
    int c = k & 63;          // input channel
    wt[i] = (_Float16)w[(o * CC + c) * KPNT + p];
}

// ---------------------------------------------------------------------------
// Kernel 2: deformable conv main. One block = 32 pixels x 64 out-channels.
//   Phase 1: bilinear gather -> f16 A_lds[m][k]   (k = p*64 + c)
//   Phase 2: 8 waves, each one 16x16 WMMA tile, 18 K-chunks of 32.
// ---------------------------------------------------------------------------
__global__ __launch_bounds__(256) void dcn_main(const float* __restrict__ x,
                                                const float* __restrict__ off,
                                                const _Float16* __restrict__ wt,
                                                const float* __restrict__ bias,
                                                float* __restrict__ out) {
    __shared__ _Float16 A_lds[TILE_M * KDIM];   // 36 KB

    const int bw = blockIdx.x & 3;              // WW / TILE_M = 4
    const int h  = (blockIdx.x >> 2) & (HH - 1);
    const int b  = blockIdx.x >> 9;
    const int w0 = bw * TILE_M;

    const int tid = threadIdx.x;
    const float* xb = x + (size_t)b * CC * HH * WW;

    // ---------------- Phase 1: bilinear sampling into LDS (f16) ------------
    for (int t = tid; t < TILE_M * KPNT; t += 256) {
        const int m = t & (TILE_M - 1);
        const int p = t >> 5;                   // TILE_M == 32
        const int w = w0 + m;

        const size_t ob = (((size_t)b * (2 * KPNT) + 2 * p) * HH + h) * WW + w;
        const float offy = off[ob];
        const float offx = off[ob + (size_t)HH * WW];

        // padded-image coords (PAD = 1); ky = p/3 - 1, kx = p%3 - 1
        const float cy = (float)(h + (p / 3)) + offy;     // h + 1 + ky
        const float cx = (float)(w + (p % 3)) + offx;     // w + 1 + kx
        const float y0f = __builtin_floorf(cy);
        const float x0f = __builtin_floorf(cx);
        const float wy1 = cy - y0f, wx1 = cx - x0f;
        const float wy0 = 1.0f - wy1, wx0 = 1.0f - wx1;

        // corner coords back in unpadded image space
        const int iy0 = (int)y0f - 1, ix0 = (int)x0f - 1;
        const int iy1 = iy0 + 1,      ix1 = ix0 + 1;
        const bool vy0 = (iy0 >= 0) && (iy0 < HH);
        const bool vy1 = (iy1 >= 0) && (iy1 < HH);
        const bool vx0 = (ix0 >= 0) && (ix0 < WW);
        const bool vx1 = (ix1 >= 0) && (ix1 < WW);

        // zero the weight of invalid corners, clamp index -> no divergent loads
        const float f00 = (vy0 && vx0) ? wy0 * wx0 : 0.0f;
        const float f01 = (vy0 && vx1) ? wy0 * wx1 : 0.0f;
        const float f10 = (vy1 && vx0) ? wy1 * wx0 : 0.0f;
        const float f11 = (vy1 && vx1) ? wy1 * wx1 : 0.0f;

        const int cy0 = iy0 < 0 ? 0 : (iy0 > HH - 1 ? HH - 1 : iy0);
        const int cy1 = iy1 < 0 ? 0 : (iy1 > HH - 1 ? HH - 1 : iy1);
        const int cx0 = ix0 < 0 ? 0 : (ix0 > WW - 1 ? WW - 1 : ix0);
        const int cx1 = ix1 < 0 ? 0 : (ix1 > WW - 1 ? WW - 1 : ix1);

        const int i00 = cy0 * WW + cx0, i01 = cy0 * WW + cx1;
        const int i10 = cy1 * WW + cx0, i11 = cy1 * WW + cx1;

        _Float16* arow = A_lds + m * KDIM + p * CC;
        const float* xc = xb;
        #pragma unroll 4
        for (int c = 0; c < CC; ++c, xc += HH * WW) {
            float v = f00 * xc[i00] + f01 * xc[i01] + f10 * xc[i10] + f11 * xc[i11];
            arow[c] = (_Float16)v;
        }
    }

    __syncthreads();

    // ---------------- Phase 2: WMMA GEMM  D[32x64] = A[32x576] * W^T -------
    const int lane  = tid & 31;
    const int wave  = tid >> 5;
    const int mtile = wave >> 2;                // 0..1 (M tiles of 16)
    const int nbase = (wave & 3) << 4;          // 0,16,32,48
    const int o     = nbase + (lane & 15);      // this lane's N (out channel)
    const int kb    = (lane >> 4) << 3;         // K-base inside chunk: 0 or 8
    const int arow  = mtile * 16 + (lane & 15); // this lane's M (pixel)

    const _Float16* aptr = A_lds + arow * KDIM + kb;
    const _Float16* bptr = wt + (size_t)o * KDIM + kb;

    v8f acc = {};
    #pragma unroll
    for (int ch = 0; ch < NCHUNK; ++ch) {
        const _Float16* ap = aptr + ch * 32;
        const _Float16* bp = bptr + ch * 32;
        v8h a0 = *(const v8h*)(ap);             // halves j   -> K = kb + j
        v8h a1 = *(const v8h*)(ap + 16);        // halves 8+j -> K = kb + 16 + j
        v8h b0 = *(const v8h*)(bp);
        v8h b1 = *(const v8h*)(bp + 16);
        v16h av = __builtin_shufflevector(a0, a1, 0, 1, 2, 3, 4, 5, 6, 7,
                                                  8, 9, 10, 11, 12, 13, 14, 15);
        v16h bv = __builtin_shufflevector(b0, b1, 0, 1, 2, 3, 4, 5, 6, 7,
                                                  8, 9, 10, 11, 12, 13, 14, 15);
        acc = __builtin_amdgcn_wmma_f32_16x16x32_f16(
            false, av, false, bv, (short)0, acc, false, false);
    }

    // D layout: VGPR r -> M = r + 8*(lane>=16), N = lane&15 => 8 contiguous w's
    const float bz = bias[o];
    const int mrow = (lane >> 4) << 3;          // 0 or 8
    float* outp = out + (((size_t)b * OO + o) * HH + h) * WW
                      + w0 + mtile * 16 + mrow;
    v8f res;
    #pragma unroll
    for (int r = 0; r < 8; ++r) res[r] = acc[r] + bz;
    *(v8f*)outp = res;                           // two global_store_b128
}

// ---------------------------------------------------------------------------
extern "C" void kernel_launch(void* const* d_in, const int* in_sizes, int n_in,
                              void* d_out, int out_size, void* d_ws, size_t ws_size,
                              hipStream_t stream) {
    const float* x      = (const float*)d_in[0];   // [8,64,128,128]
    const float* offset = (const float*)d_in[1];   // [8,18,128,128]
    const float* weight = (const float*)d_in[2];   // [64,64,3,3]
    const float* bias   = (const float*)d_in[3];   // [64]
    float* out = (float*)d_out;                    // [8,64,128,128]
    _Float16* wt = (_Float16*)d_ws;                // needs 73728 bytes

    dcn_prep_weights<<<(OO * KDIM + 255) / 256, 256, 0, stream>>>(weight, wt);

    const int grid = BB * HH * (WW / TILE_M);      // 4096 blocks
    dcn_main<<<grid, 256, 0, stream>>>(x, offset, wt, bias, out);
}
